// PreLnSelfAttention_89489938579473
// MI455X (gfx1250) — compile-verified, matrix operand order checked by reading, not run on hardware
//
#include <hip/hip_runtime.h>
#include <hip/hip_bf16.h>

// ---------------------------------------------------------------------------
// DeBERTa-style relative_key_query self-attention for MI455X (gfx1250).
// All matmuls run on v_wmma_f32_16x16x32_f16 (f16 in, fp32 accumulate).
// Workspace assumption: ws_size >= ~71 MB.
//   X16 (16MB) | Q16 (16MB) | K16 (16MB) | Vt16 (16MB, [B,H,DH,S]) |
//   E16 (0.25MB) | WTq/WTk/WTv (2MB each, pre-transposed f16 weights)
// ---------------------------------------------------------------------------

typedef __attribute__((ext_vector_type(16))) _Float16 v16h;
typedef __attribute__((ext_vector_type(8)))  _Float16 v8h;
typedef __attribute__((ext_vector_type(4)))  _Float16 v4h;
typedef __attribute__((ext_vector_type(8)))  float    v8f;

#define NB 8      // batch
#define NS 1024   // seq len
#define NH 16     // heads
#define DH 64     // head dim
#define ND 1024   // hidden = NH*DH
#define MAXP 1024 // dist_emb rows = 2*MAXP-1 = 2047

__device__ __forceinline__ v8f wmma_f16(v16h a, v16h b, v8f c) {
  return __builtin_amdgcn_wmma_f32_16x16x32_f16(
      /*neg_a=*/false, a, /*neg_b=*/false, b,
      /*c_mod=*/(short)0, c, /*reuse_a=*/false, /*reuse_b=*/false);
}

__device__ __forceinline__ v8f zero8() {
  v8f z = {0.f, 0.f, 0.f, 0.f, 0.f, 0.f, 0.f, 0.f};
  return z;
}

// Load one 16x32 (f16) A/B fragment slice for this lane.
// p points at the start of this lane's row (M for A, N for mirrored B);
// off = (lane<16) ? 0 : 8.  Reads K-chunks [off,off+8) and [16+off,24+off).
__device__ __forceinline__ v16h frag16(const _Float16* p, int off) {
  v8h c0 = *(const v8h*)(p + off);
  v8h c1 = *(const v8h*)(p + 16 + off);
  return __builtin_shufflevector(c0, c1,
      0, 1, 2, 3, 4, 5, 6, 7, 8, 9, 10, 11, 12, 13, 14, 15);
}

__device__ __forceinline__ float rmax16(float x) {
  x = fmaxf(x, __shfl_xor(x, 1));
  x = fmaxf(x, __shfl_xor(x, 2));
  x = fmaxf(x, __shfl_xor(x, 4));
  x = fmaxf(x, __shfl_xor(x, 8));
  return x;
}
__device__ __forceinline__ float rsum16(float x) {
  x += __shfl_xor(x, 1);
  x += __shfl_xor(x, 2);
  x += __shfl_xor(x, 4);
  x += __shfl_xor(x, 8);
  return x;
}

// ---------------------------------------------------------------------------
// Kernel 1: fp32 -> f16 convert (vectorized x4)
// ---------------------------------------------------------------------------
__global__ void cvt_f32_to_f16_x4(const float* __restrict__ in,
                                  _Float16* __restrict__ out, int n4) {
  int i = blockIdx.x * blockDim.x + threadIdx.x;
  if (i < n4) {
    float4 v = ((const float4*)in)[i];
    v4h o = {(_Float16)v.x, (_Float16)v.y, (_Float16)v.z, (_Float16)v.w};
    ((v4h*)out)[i] = o;
  }
}

// ---------------------------------------------------------------------------
// Kernel 2: one-time W transpose + convert: WT16[n][k] = (f16) W[k][n]
// 32x32 tiles through LDS, fully coalesced on both sides.
// ---------------------------------------------------------------------------
__global__ __launch_bounds__(256) void transpose_cvt_w(
    const float* __restrict__ W, _Float16* __restrict__ WT) {
  __shared__ float t[32][33];
  const int bx = blockIdx.x * 32;   // n base (input column)
  const int by = blockIdx.y * 32;   // k base (input row)
  const int tx = threadIdx.x & 31;
  const int ty = threadIdx.x >> 5;  // 0..7
#pragma unroll
  for (int i = 0; i < 32; i += 8)
    t[ty + i][tx] = W[(size_t)(by + ty + i) * ND + (bx + tx)];
  __syncthreads();
#pragma unroll
  for (int i = 0; i < 32; i += 8)
    WT[(size_t)(bx + ty + i) * ND + (by + tx)] = (_Float16)t[tx][ty + i];
}

// ---------------------------------------------------------------------------
// Kernel 3: projection GEMM  C[8192,1024] = X16 @ W + b, W pre-transposed f16.
// No LDS, no barriers.  Block = 256 threads (8 waves); per wave: 16 rows x
// 128 cols (8 n-tiles).  Software-pipelined: the 18 b128 loads for k+32 are
// issued (and pinned with a sched_barrier) before the 8 WMMAs on step k, so
// the WMMA block only waits on the *previous* iteration's loads.
// store_mode 0: Out f16 [B,H,S,DH];  store_mode 1: Out f16 [B,H,DH,S] (V).
// ---------------------------------------------------------------------------
__global__ __launch_bounds__(256) void qkv_proj_gemm(
    const _Float16* __restrict__ X16, const _Float16* __restrict__ WT16,
    const float* __restrict__ bias, _Float16* __restrict__ Out,
    int store_mode) {
  const int tid  = threadIdx.x;
  const int wave = tid >> 5;
  const int lane = tid & 31;
  const int hlf  = lane >> 4;
  const int l16  = lane & 15;
  const int off  = hlf * 8;

  const int m0 = blockIdx.x * 128 + wave * 16;
  const int n0 = blockIdx.y * 128;

  v8f acc[8];
#pragma unroll
  for (int nt = 0; nt < 8; ++nt) acc[nt] = zero8();

  const _Float16* Arow  = X16  + (size_t)(m0 + l16) * ND;   // A row (M=l16)
  const _Float16* Brow0 = WT16 + (size_t)(n0 + l16) * ND;   // B rows (N=l16)

  // Prologue: load k-step 0.
  v16h a_cur = frag16(Arow, off);
  v16h b_cur[8];
#pragma unroll
  for (int nt = 0; nt < 8; ++nt)
    b_cur[nt] = frag16(Brow0 + (size_t)nt * 16 * ND, off);

  for (int k0 = 0; k0 < ND; k0 += 32) {
    const int k1 = k0 + 32;
    v16h a_nxt = a_cur;
    v16h b_nxt[8];
#pragma unroll
    for (int nt = 0; nt < 8; ++nt) b_nxt[nt] = b_cur[nt];
    if (k1 < ND) {
      a_nxt = frag16(Arow + k1, off);
#pragma unroll
      for (int nt = 0; nt < 8; ++nt)
        b_nxt[nt] = frag16(Brow0 + (size_t)nt * 16 * ND + k1, off);
    }
    // Keep next-step loads above the WMMA block.
    __builtin_amdgcn_sched_barrier(0);
#pragma unroll
    for (int nt = 0; nt < 8; ++nt) acc[nt] = wmma_f16(a_cur, b_cur[nt], acc[nt]);
    a_cur = a_nxt;
#pragma unroll
    for (int nt = 0; nt < 8; ++nt) b_cur[nt] = b_nxt[nt];
  }

  // Bias + store.
#pragma unroll
  for (int nt = 0; nt < 8; ++nt) {
    int n  = n0 + nt * 16 + l16;
    int hh = n >> 6, dh = n & 63;
    float bv = bias[n];
#pragma unroll
    for (int v = 0; v < 8; ++v) {
      int m  = m0 + v + hlf * 8;                // C row: li = v + hlf*8
      int bb = m >> 10, s = m & 1023;
      _Float16 val = (_Float16)(acc[nt][v] + bv);
      if (store_mode == 0)
        Out[(((size_t)bb * NH + hh) * NS + s) * DH + dh] = val;      // [B,H,S,DH]
      else
        Out[(((size_t)bb * NH + hh) * DH + dh) * NS + s] = val;      // [B,H,DH,S]
    }
  }
}

// ---------------------------------------------------------------------------
// Kernel 4: fused attention with relative_key_query bias.
// Grid: (S/64, H, B).  Block = 128 threads = 4 waves; wave w owns 16 query
// rows l0 = bx*64 + w*16.  V is pre-transposed [B,H,DH,S] so PV B-fragments
// come straight from global; ALL LDS is wave-private -> no barriers at all.
// K fragments are loaded ONCE per chunk and used both as the B operand of
// Q@K^T and the A operand of K@E^T (identical per-lane addresses).
// sched_barriers pin the load blocks ahead of the WMMA blocks.
// Per 32-key chunk/wave:
//   scores  = Q(16x64) @ K_chunk^T          -> 4 WMMAs
//   Mq      = Q(16x64) @ E_win(48x64)^T     -> 6 WMMAs   (skew-gathered)
//   Nk      = K_chunk(32x64) @ E_win^T      -> 12 WMMAs  (skew-gathered)
//   online softmax, then ctx += P(16x32) @ V(32x64) -> 4 WMMAs
// ---------------------------------------------------------------------------
__global__ __launch_bounds__(128) void rel_attn_fused(
    const _Float16* __restrict__ q16, const _Float16* __restrict__ k16,
    const _Float16* __restrict__ vt16, const _Float16* __restrict__ E16,
    const float* __restrict__ mask, float* __restrict__ out) {
  __shared__ float    MqL[4][16 * 49];              // per-wave Q·E window
  __shared__ float    NkL[4][32 * 49];              // per-wave K·E window
  __shared__ alignas(16) _Float16 PL[4][16 * 40];   // per-wave probs (f16)

  const int tid  = threadIdx.x;
  const int wave = tid >> 5;
  const int lane = tid & 31;
  const int hlf  = lane >> 4;
  const int l16  = lane & 15;
  const int off  = hlf * 8;

  const int hh = blockIdx.y;
  const int bb = blockIdx.z;
  const int l0 = blockIdx.x * 64 + wave * 16;

  const size_t head_base = (((size_t)bb * NH + hh) * NS) * DH;  // == DH*NS too
  const _Float16* Q  = q16  + head_base;
  const _Float16* K  = k16  + head_base;
  const _Float16* Vt = vt16 + head_base;            // [DH][NS] within head

  // Q A-fragments, kept live across the whole key loop.
  const _Float16* Qrow = Q + (size_t)(l0 + l16) * DH;
  const v16h qa0 = frag16(Qrow, off);        // K-dims 0..31
  const v16h qa1 = frag16(Qrow + 32, off);   // K-dims 32..63

  float mrow[8], lrow[8];
  v8f ctx[4];
#pragma unroll
  for (int v = 0; v < 8; ++v) { mrow[v] = -1e30f; lrow[v] = 0.f; }
#pragma unroll
  for (int nt = 0; nt < 4; ++nt) ctx[nt] = zero8();

  for (int r0 = 0; r0 < NS; r0 += 32) {
    // Prefetch next chunk's streaming operands (global_prefetch_b8).
    if (r0 + 32 < NS) {
      __builtin_prefetch(K + (size_t)(r0 + 32 + l16) * DH, 0, 1);
      __builtin_prefetch(K + (size_t)(r0 + 48 + l16) * DH, 0, 1);
      __builtin_prefetch(Vt + (size_t)l16 * NS + r0 + 32, 0, 1);
      __builtin_prefetch(Vt + (size_t)(32 + l16) * NS + r0 + 32, 0, 1);
    }

    // ---- load K fragments once (serve as B for Q@K^T, A for K@E^T) ----
    v16h kf[2][2];
#pragma unroll
    for (int t = 0; t < 2; ++t) {
      const _Float16* Krow = K + (size_t)(r0 + t * 16 + l16) * DH;
      kf[t][0] = frag16(Krow, off);
      kf[t][1] = frag16(Krow + 32, off);
    }

    // ---- E window fragments: distances dmin .. dmin+47 ----
    const int dmin = l0 - r0 + 992;            // >= 0 always
    v16h ef[3][2];
#pragma unroll
    for (int nt = 0; nt < 3; ++nt) {
      int e = dmin + nt * 16 + l16;
      if (e > 2 * MAXP - 2) e = 2 * MAXP - 2;  // row 47 of window is unused
      const _Float16* Erow = E16 + (size_t)e * DH;
      ef[nt][0] = frag16(Erow, off);
      ef[nt][1] = frag16(Erow + 32, off);
    }

    // All 20 fragment loads above issue before the WMMA block below.
    __builtin_amdgcn_sched_barrier(0);

    // ---- scores = Q @ K_chunk^T ----
    v8f sc[2];
#pragma unroll
    for (int t = 0; t < 2; ++t) {
      sc[t] = zero8();
      sc[t] = wmma_f16(qa0, kf[t][0], sc[t]);
      sc[t] = wmma_f16(qa1, kf[t][1], sc[t]);
    }

    // ---- Mq = Q @ E_win^T  (16 x 48) ----
#pragma unroll
    for (int nt = 0; nt < 3; ++nt) {
      v8f mq = zero8();
      mq = wmma_f16(qa0, ef[nt][0], mq);
      mq = wmma_f16(qa1, ef[nt][1], mq);
#pragma unroll
      for (int v = 0; v < 8; ++v)
        MqL[wave][(v + hlf * 8) * 49 + nt * 16 + l16] = mq[v];
    }

    // ---- Nk = K_chunk @ E_win^T  (32 x 48), K frags reused as A ----
#pragma unroll
    for (int mt = 0; mt < 2; ++mt) {
#pragma unroll
      for (int nt = 0; nt < 3; ++nt) {
        v8f nk = zero8();
        nk = wmma_f16(kf[mt][0], ef[nt][0], nk);
        nk = wmma_f16(kf[mt][1], ef[nt][1], nk);
#pragma unroll
        for (int v = 0; v < 8; ++v)
          NkL[wave][(mt * 16 + v + hlf * 8) * 49 + nt * 16 + l16] = nk[v];
      }
    }

    // ---- V^T fragments for PV, issued early to overlap with softmax ----
    v16h vb[4];
#pragma unroll
    for (int nt = 0; nt < 4; ++nt)
      vb[nt] = frag16(Vt + (size_t)(nt * 16 + l16) * NS + r0, off);
    __builtin_amdgcn_sched_barrier(0);  // keep vb loads ahead of softmax VALU

    // ---- assemble scores: qk/8 + bias_q + bias_k + mask (skew gather) ----
    // All LDS here is wave-private; DS ops are in-order within a wave.
    float sv[2][8];
#pragma unroll
    for (int nt = 0; nt < 2; ++nt) {
      float madd = mask[(size_t)bb * NS + r0 + nt * 16 + l16];
#pragma unroll
      for (int v = 0; v < 8; ++v) {
        int li = v + hlf * 8;
        int c  = nt * 16 + l16;
        int e  = li + 31 - c;                 // 0..46
        sv[nt][v] = sc[nt][v] * 0.125f + MqL[wave][li * 49 + e] +
                    NkL[wave][c * 49 + e] + madd;
      }
    }

    // ---- online softmax (per row; row mapping matches C layout) ----
    float alpha[8];
#pragma unroll
    for (int v = 0; v < 8; ++v) {
      float mx = rmax16(fmaxf(sv[0][v], sv[1][v]));
      float mnew = fmaxf(mrow[v], mx);
      alpha[v] = __expf(mrow[v] - mnew);
      mrow[v] = mnew;
      float p0 = __expf(sv[0][v] - mnew);
      float p1 = __expf(sv[1][v] - mnew);
      sv[0][v] = p0; sv[1][v] = p1;
      lrow[v] = lrow[v] * alpha[v] + rsum16(p0 + p1);
    }

    // probs -> LDS (f16, A-operand staging) and rescale ctx
#pragma unroll
    for (int nt = 0; nt < 2; ++nt)
#pragma unroll
      for (int v = 0; v < 8; ++v)
        PL[wave][(v + hlf * 8) * 40 + nt * 16 + l16] = (_Float16)sv[nt][v];
#pragma unroll
    for (int nt = 0; nt < 4; ++nt)
#pragma unroll
      for (int v = 0; v < 8; ++v) ctx[nt][v] *= alpha[v];

    // ---- ctx += P(16x32) @ V(32x64), V^T fragments already in registers ----
    v16h pa = frag16(&PL[wave][l16 * 40], off);
#pragma unroll
    for (int nt = 0; nt < 4; ++nt) ctx[nt] = wmma_f16(pa, vb[nt], ctx[nt]);
  }

  // ---- epilogue: normalize and write [B,S,H*DH] fp32 ----
#pragma unroll
  for (int nt = 0; nt < 4; ++nt) {
#pragma unroll
    for (int v = 0; v < 8; ++v) {
      int li = v + hlf * 8;
      int d  = nt * 16 + l16;
      int l  = l0 + li;
      out[(((size_t)bb * NS + l) * NH + hh) * DH + d] = ctx[nt][v] / lrow[v];
    }
  }
}

// ---------------------------------------------------------------------------
// Host-side launcher
// ---------------------------------------------------------------------------
extern "C" void kernel_launch(void* const* d_in, const int* in_sizes, int n_in,
                              void* d_out, int out_size, void* d_ws,
                              size_t ws_size, hipStream_t stream) {
  (void)in_sizes; (void)n_in; (void)out_size; (void)ws_size;

  const float* hidden = (const float*)d_in[0];
  const float* amask  = (const float*)d_in[1];
  const float* Wq = (const float*)d_in[2];
  const float* bq = (const float*)d_in[3];
  const float* Wk = (const float*)d_in[4];
  const float* bk = (const float*)d_in[5];
  const float* Wv = (const float*)d_in[6];
  const float* bv = (const float*)d_in[7];
  const float* dist = (const float*)d_in[8];
  float* out = (float*)d_out;

  // Workspace carve-up (256B aligned).
  char* ws = (char*)d_ws;
  auto align256 = [](size_t x) { return (x + 255) & ~(size_t)255; };
  const size_t nX = (size_t)NB * NS * ND;        // 8388608
  const size_t nW = (size_t)ND * ND;             // 1048576
  const size_t nE = (size_t)(2 * MAXP - 1) * DH; // 131008
  _Float16* X16 = (_Float16*)ws; ws += align256(nX * sizeof(_Float16));
  _Float16* Q16 = (_Float16*)ws; ws += align256(nX * sizeof(_Float16));
  _Float16* K16 = (_Float16*)ws; ws += align256(nX * sizeof(_Float16));
  _Float16* Vt16 = (_Float16*)ws; ws += align256(nX * sizeof(_Float16));
  _Float16* E16 = (_Float16*)ws; ws += align256(nE * sizeof(_Float16));
  _Float16* WTq = (_Float16*)ws; ws += align256(nW * sizeof(_Float16));
  _Float16* WTk = (_Float16*)ws; ws += align256(nW * sizeof(_Float16));
  _Float16* WTv = (_Float16*)ws; ws += align256(nW * sizeof(_Float16));

  // 1) fp32 -> f16 converts
  {
    int n4 = (int)(nX / 4);
    cvt_f32_to_f16_x4<<<(n4 + 255) / 256, 256, 0, stream>>>(hidden, X16, n4);
    int e4 = (int)(nE / 4);
    cvt_f32_to_f16_x4<<<(e4 + 255) / 256, 256, 0, stream>>>(dist, E16, e4);
  }

  // 2) one-time weight transpose+convert (f16, [n][k])
  {
    dim3 grid(ND / 32, ND / 32);
    transpose_cvt_w<<<grid, 256, 0, stream>>>(Wq, WTq);
    transpose_cvt_w<<<grid, 256, 0, stream>>>(Wk, WTk);
    transpose_cvt_w<<<grid, 256, 0, stream>>>(Wv, WTv);
  }

  // 3) Q/K/V projections (pure-WMMA GEMM, software-pipelined, no LDS)
  {
    dim3 grid((NB * NS) / 128, ND / 128);
    qkv_proj_gemm<<<grid, 256, 0, stream>>>(X16, WTq, bq, Q16, 0);
    qkv_proj_gemm<<<grid, 256, 0, stream>>>(X16, WTk, bk, K16, 0);
    qkv_proj_gemm<<<grid, 256, 0, stream>>>(X16, WTv, bv, Vt16, 1);
  }

  // 4) fused attention with relative_key_query bias (barrier-free main loop)
  {
    dim3 grid(NS / 64, NH, NB);
    rel_attn_fused<<<grid, 128, 0, stream>>>(Q16, K16, Vt16, E16, amask, out);
  }
}
